// IDynamicDWConv_41394894799797
// MI455X (gfx1250) — matrix operand order repaired and checked by reading, not compiled, not run on hardware
//
#include <hip/hip_runtime.h>

typedef __bf16 bf16_t;
typedef __attribute__((ext_vector_type(16))) __bf16 v16bf;
typedef __attribute__((ext_vector_type(8)))  float  v8f;
typedef unsigned int u32x4 __attribute__((ext_vector_type(4)));
typedef int          i32x4 __attribute__((ext_vector_type(4)));
typedef int          i32x8 __attribute__((ext_vector_type(8)));

#define CCH 64

__device__ __forceinline__ int iclamp(int v, int lo, int hi) {
  return v < lo ? lo : (v > hi ? hi : v);
}

// Escape an LDS pointer + full memory clobber: tells the compiler the DMA
// (tensor_load_to_lds) may have written this memory, so subsequent ds loads
// cannot be folded to undef.
__device__ __forceinline__ void lds_publish(const void* p) {
  asm volatile("" : : "v"(p) : "memory");
}

// ---------------------------------------------------------------------------
// TDM: 1-D DMA of 36864 bf16 (73728 B) from global into LDS offset 0.
// D# per CDNA5 ISA:
//   g0: count=1 | lds_addr=0 | global_addr[56:0] | type=2
//   g1: data_size=1(2B); tensor_dim0=36864; tensor_dim1=1; tile_dim0=36864;
//       tile_dim1/2=0 (unused); tensor_dim0_stride=36864
// Issued by one wave per workgroup; completion via s_wait_tensorcnt.
// ---------------------------------------------------------------------------
__device__ __forceinline__ void tdm_load_weights(const bf16_t* src) {
  const unsigned elems = 36864u;
  unsigned long long ga = (unsigned long long)(size_t)src;
  u32x4 g0;
  g0[0] = 1u;                                                  // count=1
  g0[1] = 0u;                                                  // lds_addr
  g0[2] = (unsigned)ga;                                        // global_addr lo
  g0[3] = (unsigned)((ga >> 32) & 0x01ffffffull) | (2u << 30); // ga hi | type=2
  i32x8 g1;
  g1[0] = 0x00010000;                       // data_size = 1 -> 2 bytes
  g1[1] = (int)((elems & 0xffffu) << 16);   // tensor_dim0[15:0]
  g1[2] = (int)((elems >> 16) | (1u << 16));// tensor_dim0[31:16] | tensor_dim1=1
  g1[3] = (int)((elems & 0xffffu) << 16);   // tensor_dim1 hi=0 | tile_dim0
  g1[4] = 0;                                // tile_dim1=0, tile_dim2=0
  g1[5] = (int)elems;                       // tensor_dim0_stride lo
  g1[6] = 0;
  g1[7] = 0;
  i32x4 z4 = (i32x4)0;
  i32x8 z8 = (i32x8)0;
  __builtin_amdgcn_tensor_load_to_lds(g0, g1, z4, z4, z8, 0);
}

// ---------------------------------------------------------------------------
// Weight packing: w[L][co][ci][kh][kw] (f32) -> bf16 A-fragments.
// K ordering: k = (kh*3+kw)*64 + ci  (tap-major, 18 K-steps of 32).
// A-fragment per-lane layout (16-bit A 16x32, contiguous k order chosen here; B uses the same):
//   M = lane&15 ;  K = (lane>=16 ? 16 : 0) + e
// dst layout: [L][s][m][lane][e]
// ---------------------------------------------------------------------------
__global__ void pack_w3x3_kernel(const float* __restrict__ w, bf16_t* __restrict__ dst) {
  const int total = 3 * 18 * 4 * 32 * 16;
  int idx = blockIdx.x * blockDim.x + threadIdx.x;
  if (idx >= total) return;
  int e    = idx & 15;
  int lane = (idx >> 4) & 31;
  int m    = (idx >> 9) & 3;
  int rest = idx >> 11;
  int s    = rest % 18;
  int L    = rest / 18;
  int co = m * 16 + (lane & 15);
  int k  = s * 32 + ((lane >= 16) ? 16 : 0) + e;
  int tap = k >> 6, ci = k & 63;
  int kh = tap / 3, kw = tap - kh * 3;
  float v = w[(((size_t)(L * CCH + co) * CCH + ci) * 3 + kh) * 3 + kw];
  dst[idx] = (bf16_t)v;
}

__global__ void pack_w1x1_kernel(const float* __restrict__ w, bf16_t* __restrict__ dst) {
  const int total = 2 * 36 * 32 * 16;
  int idx = blockIdx.x * blockDim.x + threadIdx.x;
  if (idx >= total) return;
  int e    = idx & 15;
  int lane = (idx >> 4) & 31;
  int m    = (idx >> 9) % 36;
  int s    = (idx >> 9) / 36;
  int co = m * 16 + (lane & 15);
  int ci = s * 32 + ((lane >= 16) ? 16 : 0) + e;
  dst[idx] = (bf16_t)w[(size_t)co * CCH + ci];
}

// ---------------------------------------------------------------------------
// avgpool 4x4: x [2,64,256,256] (NCHW) -> channel-last [2,64,64,64] NHWC
// thread id = ((n*64+ho)*64+wo)*64+c  (c fastest -> coalesced stores)
// ---------------------------------------------------------------------------
__global__ void avgpool4_kernel(const float* __restrict__ x,
                                float* __restrict__ of, bf16_t* __restrict__ ob) {
  int idx = blockIdx.x * blockDim.x + threadIdx.x;   // 524288 exact
  int c = idx & 63, wo = (idx >> 6) & 63, ho = (idx >> 12) & 63, n = idx >> 18;
  const float* p = x + (size_t)(n * 64 + c) * 65536 + (size_t)(ho * 4) * 256 + wo * 4;
  float s = 0.f;
  #pragma unroll
  for (int r = 0; r < 4; ++r)
    #pragma unroll
    for (int q = 0; q < 4; ++q) s += p[r * 256 + q];
  float v = s * (1.f / 16.f);
  of[idx] = v;
  ob[idx] = (bf16_t)v;
}

// maxpool 2x2 on channel-last [2,64,64,64] -> channel-last [2,32,32,64]
__global__ void maxpool2_kernel(const float* __restrict__ in,
                                float* __restrict__ of, bf16_t* __restrict__ ob) {
  int idx = blockIdx.x * blockDim.x + threadIdx.x;   // 131072 exact
  int c = idx & 63, wo = (idx >> 6) & 31, ho = (idx >> 11) & 31, n = idx >> 16;
  const float* p = in + (((size_t)(n * 64 + ho * 2) * 64 + wo * 2) << 6) + c;
  float v = fmaxf(fmaxf(p[0], p[64]), fmaxf(p[4096], p[4160]));
  of[idx] = v;
  ob[idx] = (bf16_t)v;
}

// ---------------------------------------------------------------------------
// 3x3 conv, C=64->64, SAME zero pad, implicit GEMM on WMMA bf16.
// Channel-last activations: B fragment = 16 contiguous bf16 per lane
// (one v16bf load). Weights TDM-staged in LDS, A from ds_load_b128 pairs.
// One wave = 64(co) x 16(pixel) tile, 18 K-steps.
// B layout: N = lane&15 ; K = (lane>=16?16:0)+e.
// ---------------------------------------------------------------------------
__global__ void __launch_bounds__(128)
conv3x3_wmma_kernel(const bf16_t* __restrict__ in, const bf16_t* __restrict__ wp,
                    const float* __restrict__ bias, const float* __restrict__ resid,
                    float* __restrict__ outf, bf16_t* __restrict__ outb,
                    int H, int W, int leaky) {
  __shared__ bf16_t smem[36864];
  if (threadIdx.x < 32) {
    tdm_load_weights(wp);
    __builtin_amdgcn_s_wait_tensorcnt(0);
  }
  __syncthreads();
  lds_publish(smem);

  const int lane = threadIdx.x & 31;
  const int wave = blockIdx.x * (blockDim.x >> 5) + (threadIdx.x >> 5);
  const int wtiles = W >> 4;
  const int pw = wave % wtiles;
  const int h  = (wave / wtiles) % H;
  const int n  = wave / (wtiles * H);
  const int w0 = pw << 4;
  const int wpix = w0 + (lane & 15);
  const int kb = (lane & 16) ? 16 : 0;

  v8f acc[4] = {};
  const v16bf* wv = (const v16bf*)smem;

  #pragma unroll
  for (int s = 0; s < 18; ++s) {
    const int tap = s >> 1;
    const int kh = tap / 3;
    const int kw = tap - kh * 3;
    const int gh = h + kh - 1;
    const int gw = wpix + kw - 1;
    const int ci0 = ((s & 1) << 5) + kb;
    v16bf b = {};
    if ((unsigned)gh < (unsigned)H && (unsigned)gw < (unsigned)W) {
      b = *(const v16bf*)(in + (((size_t)(n * H + gh) * W + gw) << 6) + ci0);
    }
    #pragma unroll
    for (int m = 0; m < 4; ++m) {
      v16bf a = wv[(s * 4 + m) * 32 + lane];
      acc[m] = __builtin_amdgcn_wmma_f32_16x16x32_bf16(
          false, a, false, b, (short)0, acc[m], false, false);
    }
  }

  const int mrow = (lane & 16) ? 8 : 0;   // C/D layout: M = r + 8*(lane>=16)
  const size_t pixbase = ((size_t)(n * H + h) * W + (w0 + (lane & 15))) << 6;
  #pragma unroll
  for (int m = 0; m < 4; ++m) {
    #pragma unroll
    for (int r = 0; r < 8; ++r) {
      const int co = m * 16 + mrow + r;
      float v = acc[m][r] + bias[co];
      if (leaky) v = v > 0.f ? v : 0.1f * v;
      const size_t oi = pixbase + co;
      if (resid) v += resid[oi];
      if (outf) outf[oi] = v;
      outb[oi] = (bf16_t)v;
    }
  }
}

// ---------------------------------------------------------------------------
// 1x1 conv 64 -> 576 at 32x32: M=576 (36 tiles), K=64 (2 steps).
// Channel-last input (contiguous B), TDM-staged weights, NCHW f32 output
// (keeps each 32x32 wt plane contiguous for the dynconv bilinear reads).
// ---------------------------------------------------------------------------
__global__ void __launch_bounds__(128)
conv1x1_wmma_kernel(const bf16_t* __restrict__ in, const bf16_t* __restrict__ wp,
                    const float* __restrict__ bias, float* __restrict__ out) {
  __shared__ bf16_t smem[36864];
  if (threadIdx.x < 32) {
    tdm_load_weights(wp);
    __builtin_amdgcn_s_wait_tensorcnt(0);
  }
  __syncthreads();
  lds_publish(smem);

  const int lane = threadIdx.x & 31;
  const int wave = blockIdx.x * 4 + (threadIdx.x >> 5);
  const int pw = wave & 1;
  const int h  = (wave >> 1) & 31;
  const int n  = wave >> 6;
  const int w0 = pw << 4;
  const int wpix = w0 + (lane & 15);
  const int kb = (lane & 16) ? 16 : 0;

  const bf16_t* p = in + (((size_t)(n * 32 + h) * 32 + wpix) << 6);
  v16bf b0 = *(const v16bf*)(p + kb);
  v16bf b1 = *(const v16bf*)(p + 32 + kb);

  const v16bf* wv = (const v16bf*)smem;
  const int mrow = (lane & 16) ? 8 : 0;

  for (int m = 0; m < 36; ++m) {
    v8f acc = {};
    v16bf a0 = wv[(0 * 36 + m) * 32 + lane];
    v16bf a1 = wv[(1 * 36 + m) * 32 + lane];
    acc = __builtin_amdgcn_wmma_f32_16x16x32_bf16(false, a0, false, b0, (short)0, acc, false, false);
    acc = __builtin_amdgcn_wmma_f32_16x16x32_bf16(false, a1, false, b1, (short)0, acc, false, false);
    #pragma unroll
    for (int r = 0; r < 8; ++r) {
      int co = m * 16 + mrow + r;
      out[((size_t)(n * 576 + co) << 10) + h * 32 + (w0 + (lane & 15))] = acc[r] + bias[co];
    }
  }
}

// ---------------------------------------------------------------------------
// Fused bilinear x8 upsample (align_corners=False, clamped edges) +
// per-pixel dynamic depthwise 3x3 with replicate-padded x.
// wt stays at 32x32 (L2-resident); never materialize the 302MB upsample.
// ---------------------------------------------------------------------------
__global__ void dynconv_kernel(const float* __restrict__ x,
                               const float* __restrict__ wt,
                               float* __restrict__ out) {
  int idx = blockIdx.x * blockDim.x + threadIdx.x;  // 2*64*256*256 exact
  int w = idx & 255, h = (idx >> 8) & 255, c = (idx >> 16) & 63, n = idx >> 22;

  float sh = (h + 0.5f) * 0.125f - 0.5f;
  float sw = (w + 0.5f) * 0.125f - 0.5f;
  int h0 = (int)floorf(sh), w0 = (int)floorf(sw);
  float fh = sh - (float)h0, fw = sw - (float)w0;
  int h0c = iclamp(h0, 0, 31), h1c = iclamp(h0 + 1, 0, 31);
  int w0c = iclamp(w0, 0, 31), w1c = iclamp(w0 + 1, 0, 31);
  float c00 = (1.f - fh) * (1.f - fw), c01 = (1.f - fh) * fw;
  float c10 = fh * (1.f - fw),         c11 = fh * fw;

  const float* xb = x + ((size_t)(n * 64 + c) << 16);
  float acc = 0.f;
  #pragma unroll
  for (int kh = 0; kh < 3; ++kh) {
    #pragma unroll
    for (int kw = 0; kw < 3; ++kw) {
      int tap = kh * 3 + kw;
      const float* p = wt + ((size_t)(n * 576 + c * 9 + tap) << 10);
      float wv = c00 * p[h0c * 32 + w0c] + c01 * p[h0c * 32 + w1c] +
                 c10 * p[h1c * 32 + w0c] + c11 * p[h1c * 32 + w1c];
      int xh = iclamp(h + kh - 1, 0, 255);
      int xw = iclamp(w + kw - 1, 0, 255);
      acc += wv * xb[(size_t)xh * 256 + xw];
    }
  }
  out[idx] = acc;
}

// ---------------------------------------------------------------------------
extern "C" void kernel_launch(void* const* d_in, const int* in_sizes, int n_in,
                              void* d_out, int out_size, void* d_ws, size_t ws_size,
                              hipStream_t stream) {
  (void)in_sizes; (void)n_in; (void)out_size; (void)ws_size;
  const float* x    = (const float*)d_in[0];
  const float* b1w1 = (const float*)d_in[1];
  const float* b1b1 = (const float*)d_in[2];
  const float* b1w2 = (const float*)d_in[3];
  const float* b1b2 = (const float*)d_in[4];
  const float* b2w1 = (const float*)d_in[5];
  const float* b2b1 = (const float*)d_in[6];
  const float* b2w2 = (const float*)d_in[7];
  const float* b2b2 = (const float*)d_in[8];
  const float* tokw = (const float*)d_in[9];
  const float* tokb = (const float*)d_in[10];
  float* out = (float*)d_out;

  char* ws = (char*)d_ws;
  size_t off = 0;
  auto take = [&](size_t bytes) -> void* {
    void* p = ws + off;
    off += (bytes + 255) & ~(size_t)255;
    return p;
  };

  const size_t L3 = 18 * 4 * 32 * 16;  // 36864 packed elems per 3x3 layer
  bf16_t* wp3 = (bf16_t*)take(12 * L3 * sizeof(bf16_t));
  bf16_t* wp1 = (bf16_t*)take((size_t)2 * 36 * 32 * 16 * sizeof(bf16_t));
  float*  Af  = (float*) take((size_t)2 * 64 * 64 * 64 * sizeof(float));
  bf16_t* Ab  = (bf16_t*)take((size_t)2 * 64 * 64 * 64 * sizeof(bf16_t));
  bf16_t* Tb  = (bf16_t*)take((size_t)2 * 64 * 64 * 64 * sizeof(bf16_t));
  float*  Pf  = (float*) take((size_t)2 * 64 * 32 * 32 * sizeof(float));
  bf16_t* Pb  = (bf16_t*)take((size_t)2 * 64 * 32 * 32 * sizeof(bf16_t));
  float*  WT  = (float*) take((size_t)2 * 576 * 32 * 32 * sizeof(float));

  const int PK3 = 3 * (int)L3;  // 110592 elements per weight tensor
  pack_w3x3_kernel<<<(PK3 + 255) / 256, 256, 0, stream>>>(b1w1, wp3 + 0 * 3 * L3);
  pack_w3x3_kernel<<<(PK3 + 255) / 256, 256, 0, stream>>>(b1w2, wp3 + 1 * 3 * L3);
  pack_w3x3_kernel<<<(PK3 + 255) / 256, 256, 0, stream>>>(b2w1, wp3 + 2 * 3 * L3);
  pack_w3x3_kernel<<<(PK3 + 255) / 256, 256, 0, stream>>>(b2w2, wp3 + 3 * 3 * L3);
  pack_w1x1_kernel<<<(36864 + 255) / 256, 256, 0, stream>>>(tokw, wp1);

  avgpool4_kernel<<<524288 / 256, 256, 0, stream>>>(x, Af, Ab);

  for (int i = 0; i < 3; ++i) {  // block1 at 64x64: 512 wave-tiles, 4 waves/WG
    conv3x3_wmma_kernel<<<128, 128, 0, stream>>>(
        Ab, wp3 + (0 * 3 + i) * L3, b1b1 + i * 64, nullptr, nullptr, Tb, 64, 64, 1);
    conv3x3_wmma_kernel<<<128, 128, 0, stream>>>(
        Tb, wp3 + (1 * 3 + i) * L3, b1b2 + i * 64, Af, Af, Ab, 64, 64, 0);
  }

  maxpool2_kernel<<<131072 / 256, 256, 0, stream>>>(Af, Pf, Pb);

  for (int i = 0; i < 3; ++i) {  // block2 at 32x32: 128 wave-tiles
    conv3x3_wmma_kernel<<<32, 128, 0, stream>>>(
        Pb, wp3 + (2 * 3 + i) * L3, b2b1 + i * 64, nullptr, nullptr, Tb, 32, 32, 1);
    conv3x3_wmma_kernel<<<32, 128, 0, stream>>>(
        Tb, wp3 + (3 * 3 + i) * L3, b2b2 + i * 64, Pf, Pf, Pb, 32, 32, 0);
  }

  conv1x1_wmma_kernel<<<32, 128, 0, stream>>>(Pb, wp1, tokb, WT);

  dynconv_kernel<<<8388608 / 256, 256, 0, stream>>>(x, WT, out);
}
